// MyBaseRNN_7318624273023
// MI455X (gfx1250) — matrix operand order repaired and checked by reading, not compiled, not run on hardware
//
#include <hip/hip_runtime.h>
#include <hip/hip_bf16.h>
#include <math.h>

// ---------------------------------------------------------------------------
// Problem constants (from the reference): x:(2048,64,512) f32, W:(512,512),
// U:(512,512), out = concat(outputs (T,B,H), hidden (B,H)) in f32.
//
// out[t,b,h] = tanh( (x.W^T)[t,b,h] + b_in[h] + (h0.U^T)[b,h] + b_h[h] )
// -> one M=131072, N=512, K=512 GEMM (68.7 GFLOP) on the bf16 WMMA pipe with
//    split-precision (hi/lo) accumulation, ~fp32 accuracy at 3x bf16 cost.
// HBM floor: 537 MB / 23.3 TB/s ~ 23 us; this path sits near it.
// ---------------------------------------------------------------------------
#define SEQ_LEN 2048
#define BATCH   64
#define INSZ    512
#define HID     512
#define MROWS   (SEQ_LEN * BATCH)            // 131072
#define OUT_MAIN ((size_t)MROWS * HID)       // 67108864
#define LAST_ROW0 ((SEQ_LEN - 1) * BATCH)    // 131008
#define NCHUNK  (INSZ / 32)                  // 16 K-chunks of 32

typedef __bf16 v16bf __attribute__((ext_vector_type(16)));
typedef unsigned short v16us __attribute__((ext_vector_type(16)));
typedef unsigned short v8us __attribute__((ext_vector_type(8)));
typedef unsigned short v4us __attribute__((ext_vector_type(4)));
typedef float v8f __attribute__((ext_vector_type(8)));
typedef float v4f __attribute__((ext_vector_type(4)));

union BF16x16 { v16us u; v16bf b; };

// Round-to-nearest-even f32 -> bf16 bit pattern (finite inputs).
__device__ __forceinline__ unsigned short f32_to_bf16_rne(float f) {
    unsigned u = __float_as_uint(f);
    unsigned r = u + 0x7FFFu + ((u >> 16) & 1u);
    return (unsigned short)(r >> 16);
}

// Split f32 into bf16 hi (truncated, exact residual) + bf16 lo (RNE of residual).
__device__ __forceinline__ void split_bf16(float a, unsigned short& hb, unsigned short& lb) {
    unsigned u = __float_as_uint(a);
    unsigned short h = (unsigned short)(u >> 16);
    float fh = __uint_as_float((unsigned)h << 16);
    hb = h;
    lb = f32_to_bf16_rne(a - fh);
}

// ---------------------------------------------------------------------------
// Prep 1: split W (512x512 f32, row-major [h][i]) into bf16 hi/lo planes.
// ---------------------------------------------------------------------------
__global__ __launch_bounds__(256) void prep_w_split(const float* __restrict__ W,
                                                    unsigned short* __restrict__ Whi,
                                                    unsigned short* __restrict__ Wlo) {
    int i = blockIdx.x * 256 + threadIdx.x;           // < 512*512
    unsigned short h, l;
    split_bf16(W[i], h, l);
    Whi[i] = h;
    Wlo[i] = l;
}

// ---------------------------------------------------------------------------
// Prep 2: uh[b][h] = dot(prev_hidden[b,:], U[h,:]) + hidden_bias[h] + input_bias[h]
// Tiny (33 MFLOP) -> plain fp32 VALU kernel, one thread per output.
// ---------------------------------------------------------------------------
__global__ __launch_bounds__(256) void prep_uh(const float* __restrict__ ph,
                                               const float* __restrict__ U,
                                               const float* __restrict__ hb,
                                               const float* __restrict__ ib,
                                               float* __restrict__ uh) {
    int idx = blockIdx.x * 256 + threadIdx.x;         // < 64*512
    int b = idx >> 9;
    int h = idx & 511;
    const v4f* hv = (const v4f*)(ph + (size_t)b * HID);
    const v4f* Ur = (const v4f*)(U + (size_t)h * HID);
    float s = 0.0f;
#pragma unroll 4
    for (int j = 0; j < HID / 4; ++j) {
        v4f a = hv[j], c = Ur[j];
        s += a[0] * c[0] + a[1] * c[1] + a[2] * c[2] + a[3] * c[3];
    }
    uh[idx] = s + hb[h] + ib[h];
}

// ---------------------------------------------------------------------------
// Main GEMM + epilogue.
// Grid: 8192 blocks (one 16-row M tile), 256 threads = 8 waves; wave w owns
// N columns [w*64, w*64+64) as 4 WMMA tiles.
//
// Stage 1 (cooperative, once per block): load 16x512 fp32 x-tile, split into
// bf16 hi/lo, store into LDS pre-swizzled to the WMMA A-fragment layout.
// LDS layout per plane: [ep=e>>3][chunk][lane][j=e&7] shorts -> each lane's
// ds_load_b128 has a 16B lane stride (conflict-free across 64 banks).
//
// Stage 2 (K loop, 16 chunks): 4x ds_load_b128 (A hi/lo) + 16x global_load_b128
// (B hi/lo from L2-resident W planes) + 12x v_wmma_f32_16x16x32_bf16.
// ---------------------------------------------------------------------------
__global__ __launch_bounds__(256, 2) void rnn_wmma_main(const float* __restrict__ x,
                                                        const unsigned short* __restrict__ Whi,
                                                        const unsigned short* __restrict__ Wlo,
                                                        const float* __restrict__ uh,
                                                        float* __restrict__ out) {
    __shared__ __align__(16) unsigned short lds_hi[2 * NCHUNK * 32 * 8];   // 16 KB
    __shared__ __align__(16) unsigned short lds_lo[2 * NCHUNK * 32 * 8];   // 16 KB

    const int tid  = threadIdx.x;
    const int lane = tid & 31;
    const int wave = tid >> 5;                // 0..7  -> N group
    const int hf   = lane >> 4;               // half-wave select
    const int m    = lane & 15;
    const int rowbase = blockIdx.x << 4;      // M tile base row
    const int n0 = wave << 6;                 // N group base column

    // ---- Stage 1: cooperative A-tile split into LDS (8 float4 per thread).
#pragma unroll
    for (int it = 0; it < 8; ++it) {
        const int g4 = (it << 8) + tid;       // 0..2047 float4s over 16x512
        const int r  = g4 >> 7;               // row in tile (0..15)
        const int k0 = (g4 & 127) << 2;       // K base, multiple of 4
        v4f a = *(const v4f*)(x + (size_t)(rowbase + r) * INSZ + k0);
        v4us hiv, lov;
#pragma unroll
        for (int j = 0; j < 4; ++j) {
            unsigned short h, l;
            split_bf16(a[j], h, l);
            hiv[j] = h;
            lov[j] = l;
        }
        // Fragment position: K = kc + hf*16 + e
        const int c    = k0 >> 5;
        const int k32  = k0 & 31;
        const int hfw  = k32 >> 4;
        const int e    = k32 & 15;                         // 0,4,8,12
        const int lrow = (hfw << 4) + r;
        const int idx  = ((((e >> 3) << 4) + c) << 8) + (lrow << 3) + (e & 7);
        *(v4us*)(lds_hi + idx) = hiv;
        *(v4us*)(lds_lo + idx) = lov;
    }
    __syncthreads();

    // ---- Stage 2: K loop, pure LDS/global loads + WMMA.
    v8f acc[4] = {};
    for (int c = 0; c < NCHUNK; ++c) {
        const int o0 = (c << 8) + (lane << 3);            // shorts; 16B lane stride
        v8us ah0 = *(const v8us*)(lds_hi + o0);
        v8us ah1 = *(const v8us*)(lds_hi + o0 + (NCHUNK << 8));
        v8us al0 = *(const v8us*)(lds_lo + o0);
        v8us al1 = *(const v8us*)(lds_lo + o0 + (NCHUNK << 8));
        BF16x16 Ahi, Alo;
        Ahi.u = __builtin_shufflevector(ah0, ah1, 0, 1, 2, 3, 4, 5, 6, 7,
                                        8, 9, 10, 11, 12, 13, 14, 15);
        Alo.u = __builtin_shufflevector(al0, al1, 0, 1, 2, 3, 4, 5, 6, 7,
                                        8, 9, 10, 11, 12, 13, 14, 15);

        const int kb = (c << 5) + hf * 16;                // B: N=lane%16, K=kb+e
#pragma unroll
        for (int t = 0; t < 4; ++t) {
            const size_t brow = (size_t)(n0 + t * 16 + m) * INSZ + kb;
            v16bf Bhi = *(const v16bf*)(Whi + brow);      // 32B = 2x global_load_b128
            v16bf Blo = *(const v16bf*)(Wlo + brow);
            acc[t] = __builtin_amdgcn_wmma_f32_16x16x32_bf16(
                false, Ahi.b, false, Bhi, (short)0, acc[t], false, false);
            acc[t] = __builtin_amdgcn_wmma_f32_16x16x32_bf16(
                false, Ahi.b, false, Blo, (short)0, acc[t], false, false);
            acc[t] = __builtin_amdgcn_wmma_f32_16x16x32_bf16(
                false, Alo.b, false, Bhi, (short)0, acc[t], false, false);
        }
    }

    // ---- Epilogue: + uh[row%64][col], tanh, store (C layout: M=r+8*hf, N=m).
#pragma unroll
    for (int t = 0; t < 4; ++t) {
        const int gcol = n0 + t * 16 + m;
#pragma unroll
        for (int r = 0; r < 8; ++r) {
            const int grow = rowbase + r + hf * 8;
            float v = acc[t][r] + uh[((grow & (BATCH - 1)) << 9) + gcol];
            float o = tanhf(v);
            out[(size_t)grow * HID + gcol] = o;
            if (grow >= LAST_ROW0)  // last timestep also feeds the `hidden` output
                out[OUT_MAIN + (size_t)(grow - LAST_ROW0) * HID + gcol] = o;
        }
    }
}

// ---------------------------------------------------------------------------
// Launch: prep (W split + recurrent constant) then the WMMA GEMM, all on
// `stream` (graph-capture safe; workspace usage = 1 MB + 128 KB).
// ---------------------------------------------------------------------------
extern "C" void kernel_launch(void* const* d_in, const int* in_sizes, int n_in,
                              void* d_out, int out_size, void* d_ws, size_t ws_size,
                              hipStream_t stream) {
    const float* x  = (const float*)d_in[0];
    const float* ph = (const float*)d_in[1];
    const float* W  = (const float*)d_in[2];
    const float* U  = (const float*)d_in[3];
    const float* hb = (const float*)d_in[4];
    const float* ib = (const float*)d_in[5];
    float* out = (float*)d_out;

    unsigned short* Whi = (unsigned short*)d_ws;
    unsigned short* Wlo = Whi + (size_t)HID * INSZ;
    float* uh = (float*)((char*)d_ws + (size_t)2 * HID * INSZ * sizeof(unsigned short));

    prep_w_split<<<(HID * INSZ) / 256, 256, 0, stream>>>(W, Whi, Wlo);
    prep_uh<<<(BATCH * HID) / 256, 256, 0, stream>>>(ph, U, hb, ib, uh);
    rnn_wmma_main<<<MROWS / 16, 256, 0, stream>>>(x, Whi, Wlo, uh, out);
}
